// Net_42288247996849
// MI455X (gfx1250) — compile-verified, matrix operand order checked by reading, not run on hardware
//
#include <hip/hip_runtime.h>
#include <cstdint>
#include <cstddef>

#define NN 4000          // nodes
#define NE 20000         // edges
#define DD 512           // hidden dim
#define TW 4             // towers
#define FO 128           // F_OUT
#define ED 50            // edge feature dim
#define EDP 64           // padded edge dim
#define NG 32            // graphs
#define NC 10            // classes
#define NL 2             // layers
#define CDIM 2048        // TW*DD
#define AVG_DEG_LOG_F 1.2548916509f
#define EPSV 1e-5f

typedef __bf16 bf16;
typedef __attribute__((ext_vector_type(16))) __bf16 v16bf;
typedef __attribute__((ext_vector_type(8)))  float  v8f;
typedef __attribute__((ext_vector_type(8)))  unsigned int v8u;

// ---------- helpers ----------
__device__ __forceinline__ int f2ord(float f) {
  int i = __float_as_int(f);
  return (i >= 0) ? i : (i ^ 0x7fffffff);   // monotone float->int order map
}
__device__ __forceinline__ float ord2f(int i) {
  return __int_as_float((i >= 0) ? i : (i ^ 0x7fffffff));
}

__device__ __forceinline__ v8u load_frag(const bf16* p, int kadd) {
  v8u u;
#pragma unroll
  for (int v = 0; v < 8; ++v) {
    const int kp = (v < 4) ? (kadd + 2 * v) : (16 + kadd + 2 * (v - 4));
    u[v] = *(const unsigned int*)(p + kp);
  }
  return u;
}

// =====================================================================
// bf16 WMMA GEMM:  C[M x N] = A[M x K] * Bt[N x K]^T (+ bias[N])
// One wave computes a 16x64 tile (4 adjacent 16x16 N-tiles): the A
// fragment is reused across 4 v_wmma_f32_16x16x32_bf16 issues per
// K-step.  Fragment packing per CDNA5 ISA 7.12.2 (16-bit A: lanes
// 0-15 -> K {0..7,16..23}, lanes 16-31 -> K {8..15,24..31}; pairs of
// consecutive K per dword).  N must be a multiple of 64 (true at all
// call sites: 4096/512/384).  The unconditional speculative prefetch
// targets the next K-tile of the streaming A matrix; A lives inside
// the workspace arena so the address is always valid mapped memory.
// =====================================================================
__global__ __launch_bounds__(256) void gemm_bf16(
    const bf16* __restrict__ A, int lda,
    const bf16* __restrict__ Bt, int ldb,
    float* __restrict__ C, int ldc,
    const float* __restrict__ bias,
    int M, int tilesN4, int tilesTotal, int K)
{
  const int lane = threadIdx.x;                 // 0..31 (wave32)
  const int tile = blockIdx.x * 8 + threadIdx.y;
  if (tile >= tilesTotal) return;
  const int tm  = tile / tilesN4;
  const int tn4 = tile - tm * tilesN4;          // units of 64 columns
  const int half = lane >> 4;
  const int mr   = lane & 15;
  const int kadd = half << 3;
  const bf16* Ap  = A  + (size_t)(tm * 16 + mr) * lda;
  const bf16* Bp0 = Bt + (size_t)(tn4 * 64 + mr) * ldb;
  v8f acc[4] = {};
  for (int k0 = 0; k0 < K; k0 += 32) {
    __builtin_prefetch(Ap + k0 + 32, 0, 0);     // speculative, next A K-tile
    v16bf a = __builtin_bit_cast(v16bf, load_frag(Ap + k0, kadd));
#pragma unroll
    for (int j = 0; j < 4; ++j) {
      v16bf b = __builtin_bit_cast(v16bf,
          load_frag(Bp0 + (size_t)j * 16 * ldb + k0, kadd));
      acc[j] = __builtin_amdgcn_wmma_f32_16x16x32_bf16(
          false, a, false, b, (short)0, acc[j], false, false);
    }
  }
  const int rbase = tm * 16 + (half << 3);
#pragma unroll
  for (int j = 0; j < 4; ++j) {
    const int col = tn4 * 64 + j * 16 + mr;
    const float bv = bias ? bias[col] : 0.0f;
#pragma unroll
    for (int r = 0; r < 8; ++r) {
      const int row = rbase + r;
      if (row < M) C[(size_t)row * ldc + col] = acc[j][r] + bv;
    }
  }
}

// ---------- feature / weight preparation ----------
__global__ void build_x0(const float* __restrict__ nodew, const float* __restrict__ pe_w,
                         const float* __restrict__ pe_b, const float* __restrict__ acts,
                         const int* __restrict__ gidx, float* __restrict__ x,
                         bf16* __restrict__ xh)
{
  int idx = blockIdx.x * 256 + threadIdx.x;
  if (idx >= NN * DD) return;
  int n = idx >> 9, d = idx & 511;
  float v = nodew[(size_t)gidx[n] * DD + d]
          + acts[2 * n] * pe_w[d] + acts[2 * n + 1] * pe_w[DD + d] + pe_b[d];
  x[idx] = v; xh[idx] = (bf16)v;
}

__global__ void build_eah(const float* __restrict__ edge_emb_w, const int* __restrict__ sgn,
                          bf16* __restrict__ eah)
{
  int idx = blockIdx.x * 256 + threadIdx.x;
  if (idx >= NE * EDP) return;
  int e = idx >> 6, k = idx & 63;
  float v = (k < ED) ? edge_emb_w[(size_t)sgn[e] * ED + k] : 0.0f;
  eah[idx] = (bf16)v;
}

// WijT[c][k], c in [0,4096): c<2048 -> Wi (pre_w rows 0..511), else Wj (rows 512..1023)
__global__ void conv_wij(const float* __restrict__ pre_w, int l, bf16* __restrict__ WijT)
{
  int idx = blockIdx.x * 256 + threadIdx.x;
  if (idx >= 2 * CDIM * DD) return;
  int c = idx >> 9, k = idx & 511;
  int part = c >> 11, cc = c & 2047;
  int t = cc >> 9, d = cc & 511;
  int row = part * DD + k;
  float v = pre_w[((size_t)(l * TW + t) * (3 * DD) + row) * DD + d];
  WijT[(size_t)c * DD + k] = (bf16)v;
}

// CeH[k][c] = sum_f edge_enc_w[l,k,f] * We[l,t,f,d]   (folded edge weights)
__global__ void conv_ce(const float* __restrict__ eewl, const float* __restrict__ pre_w,
                        int l, bf16* __restrict__ CeH)
{
  int idx = blockIdx.x * 256 + threadIdx.x;
  if (idx >= EDP * CDIM) return;
  int k = idx >> 11, c = idx & 2047;
  float s = 0.f;
  if (k < ED) {
    int t = c >> 9, d = c & 511;
    const float* we = pre_w + ((size_t)(l * TW + t) * (3 * DD) + 2 * DD) * DD + d;
    const float* ew = eewl + (size_t)k * DD;
    for (int f = 0; f < DD; ++f) s += ew[f] * we[(size_t)f * DD];
  }
  CeH[idx] = (bf16)s;
}

// cb[c] = edge_enc_b @ We + pre_b
__global__ void conv_cb(const float* __restrict__ eebl, const float* __restrict__ pre_w,
                        const float* __restrict__ pre_b, int l, float* __restrict__ cb)
{
  int c = blockIdx.x * 256 + threadIdx.x;
  if (c >= CDIM) return;
  int t = c >> 9, d = c & 511;
  const float* we = pre_w + ((size_t)(l * TW + t) * (3 * DD) + 2 * DD) * DD + d;
  float s = pre_b[((size_t)l * TW + t) * DD + d];
  for (int f = 0; f < DD; ++f) s += eebl[f] * we[(size_t)f * DD];
  cb[c] = s;
}

__global__ void conv_wx(const float* __restrict__ post_w, int l, bf16* __restrict__ WxT)
{
  int idx = blockIdx.x * 256 + threadIdx.x;
  if (idx >= DD * DD) return;
  int q = idx >> 9, k = idx & 511;
  int t = q >> 7, o = q & 127;
  float v = post_w[((size_t)(l * TW + t) * (13 * DD) + k) * FO + o];
  WxT[(size_t)q * DD + k] = (bf16)v;
}

// WoT[t][col= kg*128+o][a] with post_w row = D + kg*4D + a
__global__ void conv_wo(const float* __restrict__ post_w, int l, bf16* __restrict__ WoT)
{
  int idx = blockIdx.x * 256 + threadIdx.x;
  if (idx >= TW * 384 * CDIM) return;
  int a = idx & 2047;
  int rest = idx >> 11;
  int col = rest % 384;
  int t = rest / 384;
  int kg = col >> 7, o = col & 127;
  int row = DD + kg * (4 * DD) + a;
  float v = post_w[((size_t)(l * TW + t) * (13 * DD) + row) * FO + o];
  WoT[((size_t)t * 384 + col) * CDIM + a] = (bf16)v;
}

__global__ void conv_lin(const float* __restrict__ lin_w, int l, bf16* __restrict__ linT)
{
  int idx = blockIdx.x * 256 + threadIdx.x;
  if (idx >= DD * DD) return;
  int col = idx >> 9, k = idx & 511;
  linT[(size_t)col * DD + k] = (bf16)lin_w[((size_t)l * DD + k) * DD + col];
}

// ---------- PNA aggregation ----------
__global__ void init_agg(float* s1, float* s2, int* mnb, int* mxb, float* cnt)
{
  int idx = blockIdx.x * 256 + threadIdx.x;
  if (idx >= NN * CDIM) return;
  s1[idx] = 0.f; s2[idx] = 0.f;
  mnb[idx] = 0x7fffffff; mxb[idx] = (int)0x80000000;
  if (idx < NN) cnt[idx] = 0.f;
}

__global__ void edge_count(const int* __restrict__ dst, float* __restrict__ cnt)
{
  int e = blockIdx.x * 256 + threadIdx.x;
  if (e < NE) atomicAdd(&cnt[dst[e]], 1.0f);
}

#define EPB 64
__global__ __launch_bounds__(256) void edge_agg(
    const float* __restrict__ P, const bf16* __restrict__ eah,
    const bf16* __restrict__ CeH, const float* __restrict__ cb,
    const int* __restrict__ src, const int* __restrict__ dst,
    float* __restrict__ s1, float* __restrict__ s2,
    int* __restrict__ mnb, int* __restrict__ mxb)
{
  __shared__ bf16  ce_s[EDP * 256];     // 32 KB
  __shared__ float ea_s[EPB][EDP];      // 16 KB
  __shared__ int   sd_s[EPB][2];
  const int tid = threadIdx.x;
  const int c0 = blockIdx.x * 256;
  for (int idx = tid; idx < EDP * 256; idx += 256) {
    int k = idx >> 8, cc = idx & 255;
    ce_s[idx] = CeH[(size_t)k * CDIM + c0 + cc];
  }
  const int e0 = blockIdx.y * EPB;
  for (int idx = tid; idx < EPB * EDP; idx += 256) {
    int ee = idx >> 6, k = idx & 63;
    int e = e0 + ee;
    ea_s[ee][k] = (e < NE) ? (float)eah[(size_t)e * EDP + k] : 0.f;
  }
  if (tid < EPB) {
    int e = e0 + tid;
    if (e < NE) { sd_s[tid][0] = src[e]; sd_s[tid][1] = dst[e]; }
  }
  __syncthreads();
  const int c = c0 + tid;
  const float cbv = cb[c];
  for (int ee = 0; ee < EPB; ++ee) {
    if (e0 + ee >= NE) break;
    const int s = sd_s[ee][0], d = sd_s[ee][1];
    float m = P[(size_t)d * (2 * CDIM) + c] + P[(size_t)s * (2 * CDIM) + CDIM + c] + cbv;
    float acc = 0.f;
#pragma unroll 8
    for (int k = 0; k < EDP; ++k) acc += ea_s[ee][k] * (float)ce_s[k * 256 + tid];
    m += acc;
    const size_t o = (size_t)d * CDIM + c;
    atomicAdd(&s1[o], m);
    atomicAdd(&s2[o], m * m);
    atomicMin(&mnb[o], f2ord(m));
    atomicMax(&mxb[o], f2ord(m));
  }
}

__global__ void node_amp(const float* __restrict__ cnt, float* __restrict__ amp)
{
  int n = blockIdx.x * 256 + threadIdx.x;
  if (n >= NN) return;
  float deg = fmaxf(cnt[n], 1.0f);
  amp[n] = logf(deg + 1.0f) / AVG_DEG_LOG_F;
}

__global__ void pna_finalize(const float* __restrict__ s1, const float* __restrict__ s2,
                             const int* __restrict__ mnb, const int* __restrict__ mxb,
                             const float* __restrict__ cnt, bf16* __restrict__ aggh)
{
  int idx = blockIdx.x * 256 + threadIdx.x;
  if (idx >= NN * CDIM) return;
  int n = idx >> 11, c = idx & 2047;
  float cn = cnt[n];
  float deg = fmaxf(cn, 1.0f);
  bool has = cn > 0.f;
  float inv = 1.0f / deg;
  float mean = s1[idx] * inv;
  float var  = s2[idx] * inv - mean * mean;
  float stdv = sqrtf(fmaxf(var, 0.f) + EPSV);
  float mnv = has ? ord2f(mnb[idx]) : 0.f;
  float mxv = has ? ord2f(mxb[idx]) : 0.f;
  int t = c >> 9, d = c & 511;
  size_t base = (size_t)n * (TW * CDIM) + (size_t)t * CDIM;
  aggh[base + d]          = (bf16)mean;
  aggh[base + 512 + d]    = (bf16)mnv;
  aggh[base + 1024 + d]   = (bf16)mxv;
  aggh[base + 1536 + d]   = (bf16)stdv;
}

// o = H1 + G1 + amp*G2 + G3/amp + post_b  -> bf16 for lin GEMM
__global__ void combine_o(const float* __restrict__ H1, const float* __restrict__ G,
                          const float* __restrict__ amp, const float* __restrict__ post_bl,
                          bf16* __restrict__ Oh)
{
  int idx = blockIdx.x * 256 + threadIdx.x;
  if (idx >= NN * DD) return;
  int n = idx >> 9, q = idx & 511;
  int t = q >> 7, o = q & 127;
  float a = amp[n];
  const float* g = G + (size_t)n * (TW * 384) + (size_t)t * 384;
  float v = H1[idx] + g[o] + a * g[128 + o] + g[256 + o] / a + post_bl[t * FO + o];
  Oh[idx] = (bf16)v;
}

// ---------- batch norm ----------
__global__ __launch_bounds__(256) void bn_stats(const float* __restrict__ h,
                                                float* __restrict__ mu, float* __restrict__ var)
{
  int c = blockIdx.x;
  float s = 0.f, q = 0.f;
  for (int n = threadIdx.x; n < NN; n += 256) {
    float v = h[(size_t)n * DD + c];
    s += v; q += v * v;
  }
  __shared__ float ss[256], qq[256];
  ss[threadIdx.x] = s; qq[threadIdx.x] = q;
  __syncthreads();
  for (int off = 128; off > 0; off >>= 1) {
    if (threadIdx.x < off) { ss[threadIdx.x] += ss[threadIdx.x + off]; qq[threadIdx.x] += qq[threadIdx.x + off]; }
    __syncthreads();
  }
  if (threadIdx.x == 0) {
    float m = ss[0] / (float)NN;
    mu[c] = m;
    var[c] = qq[0] / (float)NN - m * m;
  }
}

__global__ void bn_apply(const float* __restrict__ h, const float* __restrict__ mu,
                         const float* __restrict__ var, const float* __restrict__ gamma,
                         const float* __restrict__ beta, float* __restrict__ x,
                         bf16* __restrict__ xh)
{
  int idx = blockIdx.x * 256 + threadIdx.x;
  if (idx >= NN * DD) return;
  int c = idx & 511;
  float v = gamma[c] * (h[idx] - mu[c]) * rsqrtf(var[c] + EPSV) + beta[c];
  v = fmaxf(v, 0.f);
  x[idx] = v; xh[idx] = (bf16)v;
}

// ---------- pooling + head ----------
__global__ void zero_f(float* p, int total)
{
  int idx = blockIdx.x * 256 + threadIdx.x;
  if (idx < total) p[idx] = 0.f;
}

__global__ void pool_sum(const float* __restrict__ x, const int* __restrict__ batch,
                         float* __restrict__ pooled)
{
  int idx = blockIdx.x * 256 + threadIdx.x;
  if (idx >= NN * DD) return;
  int n = idx >> 9, c = idx & 511;
  atomicAdd(&pooled[(size_t)batch[n] * DD + c], x[idx]);
}

__global__ __launch_bounds__(256) void head_fc1(const float* __restrict__ pooled,
                                                const float* __restrict__ w,
                                                const float* __restrict__ b,
                                                const float* __restrict__ prelu_a,
                                                float* __restrict__ z)
{
  int g = blockIdx.x;
  float a = *prelu_a;
  for (int j = threadIdx.x; j < 2 * DD; j += 256) {
    float acc = b[j];
    for (int k = 0; k < DD; ++k) acc += pooled[(size_t)g * DD + k] * w[(size_t)k * (2 * DD) + j];
    z[(size_t)g * (2 * DD) + j] = acc > 0.f ? acc : a * acc;
  }
}

__global__ void head_out(const float* __restrict__ z, const float* __restrict__ w,
                         const float* __restrict__ b, float* __restrict__ out)
{
  int g = blockIdx.x;
  __shared__ float lg[16];
  __shared__ float mx_s, lse_s;
  int c = threadIdx.x;
  if (c < NC) {
    float acc = b[c];
    for (int j = 0; j < 2 * DD; ++j) acc += z[(size_t)g * (2 * DD) + j] * w[(size_t)j * NC + c];
    lg[c] = acc;
  }
  __syncthreads();
  if (c == 0) {
    float m = -1e30f;
    for (int i = 0; i < NC; ++i) m = fmaxf(m, lg[i]);
    float s = 0.f;
    for (int i = 0; i < NC; ++i) s += expf(lg[i] - m);
    mx_s = m; lse_s = logf(s);
  }
  __syncthreads();
  if (c < NC) out[(size_t)g * NC + c] = lg[c] - mx_s - lse_s;
}

// ---------- host side ----------
static inline void launch_gemm(hipStream_t st, const bf16* A, int lda, const bf16* Bt, int ldb,
                               float* C, int ldc, const float* bias, int M, int N, int K)
{
  int tm = (M + 15) / 16, tn4 = N / 64;   // N multiple of 64 at all call sites
  int tiles = tm * tn4;
  dim3 grid((tiles + 7) / 8), blk(32, 8);
  gemm_bf16<<<grid, blk, 0, st>>>(A, lda, Bt, ldb, C, ldc, bias, M, tn4, tiles, K);
}

extern "C" void kernel_launch(void* const* d_in, const int* in_sizes, int n_in,
                              void* d_out, int out_size, void* d_ws, size_t ws_size,
                              hipStream_t stream)
{
  (void)in_sizes; (void)n_in; (void)out_size; (void)ws_size;
  const float* node_emb_w = (const float*)d_in[0];
  const float* edge_emb_w = (const float*)d_in[1];
  const float* pe_w      = (const float*)d_in[2];
  const float* pe_b      = (const float*)d_in[3];
  const float* edge_enc_w= (const float*)d_in[4];
  const float* edge_enc_b= (const float*)d_in[5];
  const float* pre_w     = (const float*)d_in[6];
  const float* pre_b     = (const float*)d_in[7];
  const float* post_w    = (const float*)d_in[8];
  const float* post_b    = (const float*)d_in[9];
  const float* lin_w     = (const float*)d_in[10];
  const float* lin_b     = (const float*)d_in[11];
  const float* bn_gamma  = (const float*)d_in[12];
  const float* bn_beta   = (const float*)d_in[13];
  const float* acts      = (const float*)d_in[14];
  const float* fc1_w     = (const float*)d_in[15];
  const float* fc1_b     = (const float*)d_in[16];
  const float* fc_out_w  = (const float*)d_in[17];
  const float* fc_out_b  = (const float*)d_in[18];
  const float* prelu_a   = (const float*)d_in[19];
  const int* gidx  = (const int*)d_in[20];
  const int* sgn   = (const int*)d_in[21];
  const int* eidx  = (const int*)d_in[22];
  const int* batch = (const int*)d_in[23];
  const int* src = eidx;
  const int* dst = eidx + NE;

  uint8_t* wp = (uint8_t*)d_ws;
  auto take = [&](size_t bytes) -> void* {
    void* p = (void*)wp;
    wp += (bytes + 255) & ~(size_t)255;
    return p;
  };
  float* x    = (float*)take((size_t)NN * DD * 4);
  bf16*  Xh   = (bf16*) take((size_t)NN * DD * 2);
  float* P    = (float*)take((size_t)NN * 2 * CDIM * 4);
  float* s1   = (float*)take((size_t)NN * CDIM * 4);
  float* s2   = (float*)take((size_t)NN * CDIM * 4);
  int*   mnb  = (int*)  take((size_t)NN * CDIM * 4);
  int*   mxb  = (int*)  take((size_t)NN * CDIM * 4);
  float* cnt  = (float*)take((size_t)NN * 4);
  float* amp  = (float*)take((size_t)NN * 4);
  bf16*  AGGh = (bf16*) take((size_t)NN * TW * CDIM * 2);
  float* H1   = (float*)take((size_t)NN * DD * 4);
  float* G    = (float*)take((size_t)NN * TW * 384 * 4);
  bf16*  Oh   = (bf16*) take((size_t)NN * DD * 2);
  float* hbuf = (float*)take((size_t)NN * DD * 4);
  float* mu   = (float*)take((size_t)DD * 4);
  float* var  = (float*)take((size_t)DD * 4);
  bf16*  eah  = (bf16*) take((size_t)NE * EDP * 2);
  bf16*  CeH  = (bf16*) take((size_t)EDP * CDIM * 2);
  float* cb   = (float*)take((size_t)CDIM * 4);
  bf16*  WijT = (bf16*) take((size_t)2 * CDIM * DD * 2);
  bf16*  WxT  = (bf16*) take((size_t)DD * DD * 2);
  bf16*  WoT  = (bf16*) take((size_t)TW * 384 * CDIM * 2);
  bf16*  linT = (bf16*) take((size_t)DD * DD * 2);
  float* pooled = (float*)take((size_t)NG * DD * 4);
  float* z    = (float*)take((size_t)NG * 2 * DD * 4);

  const int T256 = 256;
  build_x0<<<(NN * DD + 255) / 256, T256, 0, stream>>>(node_emb_w, pe_w, pe_b, acts, gidx, x, Xh);
  build_eah<<<(NE * EDP + 255) / 256, T256, 0, stream>>>(edge_emb_w, sgn, eah);

  for (int l = 0; l < NL; ++l) {
    conv_wij<<<(2 * CDIM * DD + 255) / 256, T256, 0, stream>>>(pre_w, l, WijT);
    conv_ce <<<(EDP * CDIM + 255) / 256, T256, 0, stream>>>(edge_enc_w + (size_t)l * ED * DD, pre_w, l, CeH);
    conv_cb <<<(CDIM + 255) / 256, T256, 0, stream>>>(edge_enc_b + (size_t)l * DD, pre_w, pre_b, l, cb);
    conv_wx <<<(DD * DD + 255) / 256, T256, 0, stream>>>(post_w, l, WxT);
    conv_wo <<<(TW * 384 * CDIM + 255) / 256, T256, 0, stream>>>(post_w, l, WoT);
    conv_lin<<<(DD * DD + 255) / 256, T256, 0, stream>>>(lin_w, l, linT);

    // P = [X@Wi | X@Wj]  (4000 x 4096, K=512) via WMMA
    launch_gemm(stream, Xh, DD, WijT, DD, P, 2 * CDIM, nullptr, NN, 2 * CDIM, DD);

    init_agg<<<(NN * CDIM + 255) / 256, T256, 0, stream>>>(s1, s2, mnb, mxb, cnt);
    edge_count<<<(NE + 255) / 256, T256, 0, stream>>>(dst, cnt);
    edge_agg<<<dim3(CDIM / 256, (NE + EPB - 1) / EPB), T256, 0, stream>>>(
        P, eah, CeH, cb, src, dst, s1, s2, mnb, mxb);
    node_amp<<<(NN + 255) / 256, T256, 0, stream>>>(cnt, amp);
    pna_finalize<<<(NN * CDIM + 255) / 256, T256, 0, stream>>>(s1, s2, mnb, mxb, cnt, AGGh);

    // H1 = X @ Wx_cat  (4000 x 512, K=512)
    launch_gemm(stream, Xh, DD, WxT, DD, H1, DD, nullptr, NN, DD, DD);
    // G[t] = AGG_t @ [Wo1|Wo2|Wo3]_t  (4000 x 384, K=2048) per tower
    for (int t = 0; t < TW; ++t)
      launch_gemm(stream, AGGh + (size_t)t * CDIM, TW * CDIM,
                  WoT + (size_t)t * 384 * CDIM, CDIM,
                  G + (size_t)t * 384, TW * 384, nullptr, NN, 384, CDIM);
    combine_o<<<(NN * DD + 255) / 256, T256, 0, stream>>>(
        H1, G, amp, post_b + (size_t)l * TW * FO, Oh);
    // h = O @ lin_w + lin_b  (4000 x 512, K=512)
    launch_gemm(stream, Oh, DD, linT, DD, hbuf, DD, lin_b + (size_t)l * DD, NN, DD, DD);

    bn_stats<<<DD, T256, 0, stream>>>(hbuf, mu, var);
    bn_apply<<<(NN * DD + 255) / 256, T256, 0, stream>>>(
        hbuf, mu, var, bn_gamma + (size_t)l * DD, bn_beta + (size_t)l * DD, x, Xh);
  }

  zero_f<<<(NG * DD + 255) / 256, T256, 0, stream>>>(pooled, NG * DD);
  pool_sum<<<(NN * DD + 255) / 256, T256, 0, stream>>>(x, batch, pooled);
  head_fc1<<<NG, T256, 0, stream>>>(pooled, fc1_w, fc1_b, prelu_a, z);
  head_out<<<NG, 32, 0, stream>>>(z, fc_out_w, fc_out_b, (float*)d_out);
}
